// Attention_89240830476455
// MI455X (gfx1250) — matrix operand order repaired and checked by reading, not compiled, not run on hardware
//
#include <hip/hip_runtime.h>
#include <hip/hip_bf16.h>

// ---------------------------------------------------------------------------
// CDNA5 (gfx1250) windowed channel attention block.
// - all contractions via v_wmma_f32_16x16x32_f16 (wave32)
// - f16 intermediates in workspace (halves HBM traffic; pipeline is BW-bound)
// - attention q/k gather via global_load_async_to_lds_b64/b128 + s_wait_asynccnt
// ---------------------------------------------------------------------------

typedef _Float16 v16h __attribute__((ext_vector_type(16)));
typedef _Float16 v8h  __attribute__((ext_vector_type(8)));
typedef float    v8f  __attribute__((ext_vector_type(8)));

#define HWDIM 65536          // 256*256
#define BATCH 4
#define KDIM  144
#define KPAD  160            // K rounded to 32 for the WMMA loop
#define KP    168            // LDS pitch (halves) for GEMM X tile

__device__ __forceinline__ v16h cat8(v8h a, v8h b) {
    return __builtin_shufflevector(a, b, 0,1,2,3,4,5,6,7,8,9,10,11,12,13,14,15);
}

// A-fragment (16x32 f16, ISA 7.12.2): lane row = lane%16,
// elems 0..7 -> k0 + (lane/16)*8 .. +7 ; elems 8..15 -> k0+16+(lane/16)*8 .. +7
__device__ __forceinline__ v16h ldsA(const _Float16* base, int pitch, int row,
                                     int k0, int lane) {
    const _Float16* p = base + row * pitch + k0 + (lane >> 4) * 8;
    v8h lo = *(const v8h*)p;
    v8h hi = *(const v8h*)(p + 16);
    return cat8(lo, hi);
}

// B-fragment (32x16 f16): lane col = lane%16, elems 0..7 -> k0 + (lane/16)*8 .. +7 ;
// elems 8..15 -> k0+16+(lane/16)*8 .. +7 ; buffer stored as [n][k].
__device__ __forceinline__ v16h ldsB(const _Float16* base, int pitch, int row,
                                     int k0, int lane) {
    const _Float16* p = base + row * pitch + k0 + (lane >> 4) * 8;
    v8h lo = *(const v8h*)p;
    v8h hi = *(const v8h*)(p + 16);
    return cat8(lo, hi);
}

// A-fragment from a row-major f16 weight row (L2-resident). K=144: for
// k0=128 the hi half (k>=144) is fully OOB -> compile-time zero.
__device__ __forceinline__ v16h gAh(const _Float16* __restrict__ row, int k0,
                                    int lane) {
    int g = (lane >> 4) * 8;
    v8h lo = *(const v8h*)(row + k0 + g);
    v8h hi = {};
    if (k0 + 16 < KDIM) hi = *(const v8h*)(row + k0 + 16 + g);
    return cat8(lo, hi);
}

__device__ __forceinline__ v8f wmma_f16(v16h a, v16h b, v8f c) {
    return __builtin_amdgcn_wmma_f32_16x16x32_f16(false, a, false, b,
                                                  (short)0, c, false, false);
}

// --- CDNA5 async global->LDS copies (ASYNCcnt-tracked, ISA ch.10/15.18) ----
__device__ __forceinline__ void async_g2l_b64(void* lds, const void* g) {
    asm volatile("global_load_async_to_lds_b64 %0, %1, off"
                 :: "v"((uint32_t)(uintptr_t)lds), "v"(g) : "memory");
}
__device__ __forceinline__ void async_g2l_b128(void* lds, const void* g) {
    asm volatile("global_load_async_to_lds_b128 %0, %1, off"
                 :: "v"((uint32_t)(uintptr_t)lds), "v"(g) : "memory");
}
__device__ __forceinline__ void wait_async0() {
    asm volatile("s_wait_asynccnt 0x0" ::: "memory");
}

// ---------------------------------------------------------------------------
// One-shot f32 -> f16 conversion (weights)
// ---------------------------------------------------------------------------
__global__ __launch_bounds__(256) void f32to16(const float* __restrict__ in,
                                               _Float16* __restrict__ out,
                                               int n) {
    int i = blockIdx.x * 256 + threadIdx.x;
    if (i < n) out[i] = (_Float16)in[i];
}

// ---------------------------------------------------------------------------
// 1x1 conv as GEMM: out[b,o,p] = sum_c w[o,c]*in[b,c,p] + bias[o]
// Block: 256 thr (8 waves). 16-column X tile staged once in LDS (as f16,
// [n][k] transposed); each wave sweeps M-tiles -> activations read once.
// TIN: float (stage 1) or f16 (proj). TOUT: f16 (stage 1) or float (final).
// ---------------------------------------------------------------------------
template <typename TIN, typename TOUT>
__global__ __launch_bounds__(256) void gemm1x1(const TIN* __restrict__ in,
                                               const _Float16* __restrict__ wgt,
                                               const float* __restrict__ bias,
                                               TOUT* __restrict__ out, int M) {
    const int b  = blockIdx.y;
    const int p0 = blockIdx.x * 16;
    const TIN* inb = in + (size_t)b * KDIM * HWDIM;
    TOUT* outb     = out + (size_t)b * M * HWDIM;

    __shared__ _Float16 xs[16 * KP];

    const int tid = threadIdx.x;
    if constexpr (sizeof(TIN) == 4) {
        // f32 input: thread loads float4 along n, converts, scatters to [n][k]
#pragma unroll
        for (int i = 0; i < 3; ++i) {            // 160 k * 4 quads = 640 units
            int idx = tid + i * 256;
            if (idx < 640) {
                int k = idx >> 2, n4 = (idx & 3) * 4;
                float4 f = make_float4(0.f, 0.f, 0.f, 0.f);
                if (k < KDIM)
                    f = *(const float4*)((const float*)inb + (size_t)k * HWDIM + p0 + n4);
                xs[(n4 + 0) * KP + k] = (_Float16)f.x;
                xs[(n4 + 1) * KP + k] = (_Float16)f.y;
                xs[(n4 + 2) * KP + k] = (_Float16)f.z;
                xs[(n4 + 3) * KP + k] = (_Float16)f.w;
            }
        }
    } else {
        // f16 input: thread loads a dword (2 halves along n)
#pragma unroll
        for (int i = 0; i < 5; ++i) {            // 160 k * 8 pairs = 1280 units
            int idx = tid + i * 256;
            int k = idx >> 3, n2 = (idx & 7) * 2;
            union { uint32_t u; _Float16 h[2]; } uu;
            uu.u = 0;
            if (k < KDIM)
                uu.u = *(const uint32_t*)((const _Float16*)inb + (size_t)k * HWDIM + p0 + n2);
            xs[(n2 + 0) * KP + k] = uu.h[0];
            xs[(n2 + 1) * KP + k] = uu.h[1];
        }
    }
    __syncthreads();

    const int lane = tid & 31;
    const int wave = tid >> 5;
    const int n = lane & 15;
    const int g = lane >> 4;

    for (int mt = wave; mt * 16 < M; mt += 8) {
        const int mrow = mt * 16 + (lane & 15);
        const _Float16* wrow = wgt + (size_t)mrow * KDIM;
        v8f acc = {};
#pragma unroll
        for (int k0 = 0; k0 < KPAD; k0 += 32) {
            v16h a  = gAh(wrow, k0, lane);
            v16h bb = ldsB(xs, KP, n, k0, lane);
            acc = wmma_f16(a, bb, acc);
        }
#pragma unroll
        for (int r = 0; r < 8; ++r) {
            int o = mt * 16 + r + 8 * g;      // C/D layout: M = r + 8*(lane/16)
            outb[(size_t)o * HWDIM + p0 + n] = (TOUT)(acc[r] + bias[o]);
        }
    }
}

// ---------------------------------------------------------------------------
// Depthwise 3x3, SAME padding, f16 in/out, f32 accumulate. Pure BW stage.
// ---------------------------------------------------------------------------
__global__ __launch_bounds__(256) void dwconv3x3(const _Float16* __restrict__ in,
                                                 const float* __restrict__ wgt,
                                                 const float* __restrict__ bias,
                                                 _Float16* __restrict__ out) {
    const int bc = blockIdx.y;                // b*432 + c
    const int c  = bc % 432;
    const int p  = blockIdx.x * 256 + threadIdx.x;
    const int h = p >> 8, w = p & 255;
    const _Float16* src = in + (size_t)bc * HWDIM;
    float acc = bias[c];
#pragma unroll
    for (int i = 0; i < 3; ++i)
#pragma unroll
        for (int j = 0; j < 3; ++j) {
            int hh = h + i - 1, ww = w + j - 1;
            if (hh >= 0 && hh < 256 && ww >= 0 && ww < 256)
                acc += (float)src[hh * 256 + ww] * wgt[c * 9 + i * 3 + j];
        }
    out[(size_t)bc * HWDIM + p] = (_Float16)acc;
}

// ---------------------------------------------------------------------------
// Windowed channel attention, one branch. 288-thr block (9 wave32) / window.
//   q,k rows async-copied global->LDS (b64/b128, ASYNCcnt), v transposed,
//   l2norm via ds_add_f32, S = qn*kn^T (9 WMMA tiles), f32 softmax,
//   O = S*v (WMMA, K 48->64 zero-padded), f16 scatter to unwindowed layout.
// ---------------------------------------------------------------------------
template <int BR, int DH, int DW>
__global__ __launch_bounds__(288) void winattn(const _Float16* __restrict__ y,
                                               _Float16* __restrict__ out) {
    constexpr int D    = DH * DW;
    constexpr int DPAD = (D < 32) ? 32 : D;
    constexpr int QKP  = DPAD + 8;            // pitch (halves), multiple of 8
    constexpr int VP   = 72;                  // vT pitch: 64 pad + 8
    constexpr int SHP  = 72;                  // Sh pitch
    constexpr int SP   = 52;                  // Sf pitch (floats)

    extern __shared__ float4 smem4[];
    _Float16* qs = (_Float16*)smem4;          // 48*QKP
    _Float16* ks = qs + 48 * QKP;             // 48*QKP
    _Float16* vT = ks + 48 * QKP;             // D*VP   (spatial-major, c padded)
    float*    Sf = (float*)(vT + D * VP);     // 48*SP
    _Float16* Sh = (_Float16*)(Sf + 48 * SP); // 48*SHP
    __shared__ float nrm[96];

    const int tid  = threadIdx.x;
    const int b    = blockIdx.y;
    const int win  = blockIdx.x;
    const int h0   = (win / (256 / DW)) * DH;
    const int w0   = (win % (256 / DW)) * DW;
    const _Float16* base = y + ((size_t)b * 432 + BR * 144) * HWDIM;

    // 1) issue async q/k row copies: each unit = one contiguous window row
    //    (DW halves) for one channel; both global & LDS addresses contiguous.
    constexpr int SUB = (DW == 16) ? 2 : 1;   // 32B rows need two b128s
    for (int i = tid; i < 48 * DH * 2 * SUB; i += 288) {
        int s   = i % SUB;
        int t2  = i / SUB;
        int isk = t2 & 1;
        int rc  = t2 >> 1;
        int c = rc / DH, wr = rc % DH;
        const _Float16* gp = base + (size_t)(isk ? 48 + c : c) * HWDIM
                           + (size_t)(h0 + wr) * 256 + w0 + s * 8;
        _Float16* lp = (isk ? ks : qs) + c * QKP + wr * DW + s * 8;
        if (DW == 4) async_g2l_b64(lp, gp);
        else         async_g2l_b128(lp, gp);
    }

    // 2) pad-region zero fills (disjoint from async-written regions)
    if (tid < 96) nrm[tid] = 0.0f;
    if (D == 16) {                            // qs/ks cols [16,32)
        for (int i = tid; i < 48 * 16; i += 288) {
            int c = i >> 4, j = 16 + (i & 15);
            qs[c * QKP + j] = (_Float16)0;
            ks[c * QKP + j] = (_Float16)0;
        }
    }
    for (int i = tid; i < D * 16; i += 288)   // vT cols [48,64)
        vT[(i >> 4) * VP + 48 + (i & 15)] = (_Float16)0;
    for (int i = tid; i < 48 * 16; i += 288)  // Sh cols [48,64)
        Sh[(i >> 4) * SHP + 48 + (i & 15)] = (_Float16)0;

    // 3) gather v (transposed store; f16 copy, no cvt)
    for (int i = tid; i < 48 * D; i += 288) {
        int c = i / D, j = i % D;
        int r = j / DW, cc = j % DW;
        size_t sp = (size_t)(h0 + r) * 256 + (w0 + cc);
        vT[j * VP + c] = base[(size_t)(96 + c) * HWDIM + sp];
    }

    wait_async0();
    __syncthreads();

    // 4) row sum-of-squares from LDS (ds_add_f32 atomics)
    for (int i = tid; i < 48 * D; i += 288) {
        int c = i / D, j = i % D;
        float qv = (float)qs[c * QKP + j];
        float kv = (float)ks[c * QKP + j];
        atomicAdd(&nrm[c], qv * qv);
        atomicAdd(&nrm[48 + c], kv * kv);
    }
    __syncthreads();

    // 5) l2 normalize rows of q,k (eps per torch F.normalize)
    for (int i = tid; i < 48 * D; i += 288) {
        int c = i / D, j = i % D;
        float qsc = 1.0f / fmaxf(sqrtf(nrm[c]), 1e-12f);
        float ksc = 1.0f / fmaxf(sqrtf(nrm[48 + c]), 1e-12f);
        qs[c * QKP + j] = (_Float16)((float)qs[c * QKP + j] * qsc);
        ks[c * QKP + j] = (_Float16)((float)ks[c * QKP + j] * ksc);
    }
    __syncthreads();

    const int lane = tid & 31;
    const int wave = tid >> 5;                // 0..8
    const int g = lane >> 4;

    // 6) S = qn * kn^T : 9 16x16 tiles, one per wave
    {
        int mt = wave / 3, nt = wave % 3;
        int mrow = mt * 16 + (lane & 15);
        int nrow = nt * 16 + (lane & 15);
        v8f acc = {};
#pragma unroll
        for (int k0 = 0; k0 < DPAD; k0 += 32) {
            v16h a  = ldsA(qs, QKP, mrow, k0, lane);
            v16h bb = ldsB(ks, QKP, nrow, k0, lane);
            acc = wmma_f16(a, bb, acc);
        }
#pragma unroll
        for (int r = 0; r < 8; ++r)
            Sf[(mt * 16 + r + 8 * g) * SP + nt * 16 + (lane & 15)] = acc[r];
    }
    __syncthreads();

    // 7) f32 softmax over rows of S, store as f16 (cols 48..63 stay zero)
    if (tid < 48) {
        float mx = -1e30f;
        for (int j = 0; j < 48; ++j) mx = fmaxf(mx, Sf[tid * SP + j]);
        float s = 0.0f;
        for (int j = 0; j < 48; ++j) s += __expf(Sf[tid * SP + j] - mx);
        float inv = 1.0f / s;
        for (int j = 0; j < 48; ++j)
            Sh[tid * SHP + j] = (_Float16)(__expf(Sf[tid * SP + j] - mx) * inv);
    }
    __syncthreads();

    // 8) O = Sh(48x64pad) * v(64pad x D), f16 scatter to unwindowed layout
    _Float16* ob = out + ((size_t)b * 144 + BR * 48) * HWDIM;
    constexpr int NT = D / 16;
    for (int t = wave; t < 3 * NT; t += 9) {
        int mt = t % 3, nt = t / 3;
        int mrow = mt * 16 + (lane & 15);
        int nrow = nt * 16 + (lane & 15);
        v8f acc = {};
#pragma unroll
        for (int k0 = 0; k0 < 64; k0 += 32) {
            v16h a  = ldsA(Sh, SHP, mrow, k0, lane);
            v16h bb = ldsB(vT, VP, nrow, k0, lane);
            acc = wmma_f16(a, bb, acc);
        }
        int j  = nt * 16 + (lane & 15);
        int rr = j / DW, cc = j % DW;
        size_t sp = (size_t)(h0 + rr) * 256 + (w0 + cc);
#pragma unroll
        for (int r = 0; r < 8; ++r) {
            int c = mt * 16 + r + 8 * g;
            ob[(size_t)c * HWDIM + sp] = (_Float16)acc[r];
        }
    }
}

// ---------------------------------------------------------------------------
// Host launcher
// ---------------------------------------------------------------------------
extern "C" void kernel_launch(void* const* d_in, const int* in_sizes, int n_in,
                              void* d_out, int out_size, void* d_ws,
                              size_t ws_size, hipStream_t stream) {
    (void)in_sizes; (void)n_in; (void)out_size; (void)ws_size;

    const float* x      = (const float*)d_in[0];
    const float* qkv_w  = (const float*)d_in[1];
    const float* qkv_b  = (const float*)d_in[2];
    const float* dw_w   = (const float*)d_in[3];
    const float* dw_b   = (const float*)d_in[4];
    const float* proj_w = (const float*)d_in[5];
    const float* proj_b = (const float*)d_in[6];
    float* outp = (float*)d_out;

    // f16 workspace layout
    _Float16* y1  = (_Float16*)d_ws;                        // (4,432,256,256)
    _Float16* y2  = y1 + (size_t)BATCH * 432 * HWDIM;       // (4,432,256,256)
    _Float16* ya  = y2 + (size_t)BATCH * 432 * HWDIM;       // (4,144,256,256)
    _Float16* qwh = ya + (size_t)BATCH * 144 * HWDIM;       // 432*144
    _Float16* pwh = qwh + 432 * KDIM;                       // 144*144

    // dynamic LDS per branch
    auto smemsz = [](int D, int QKP) {
        return (size_t)(2 * 48 * QKP + D * 72 + 48 * 72) * 2 + 48 * 52 * 4;
    };
    const size_t s0 = smemsz(16, 40);
    const size_t s1 = smemsz(64, 72);
    const size_t s2 = smemsz(256, 264);
    hipFuncSetAttribute((const void*)winattn<0, 4, 4>,
                        hipFuncAttributeMaxDynamicSharedMemorySize, (int)s0);
    hipFuncSetAttribute((const void*)winattn<1, 8, 8>,
                        hipFuncAttributeMaxDynamicSharedMemorySize, (int)s1);
    hipFuncSetAttribute((const void*)winattn<2, 16, 16>,
                        hipFuncAttributeMaxDynamicSharedMemorySize, (int)s2);

    // 0) weights -> f16 once
    f32to16<<<dim3((432 * KDIM + 255) / 256), 256, 0, stream>>>(qkv_w, qwh, 432 * KDIM);
    f32to16<<<dim3((144 * KDIM + 255) / 256), 256, 0, stream>>>(proj_w, pwh, 144 * KDIM);

    // 1) qkv 1x1 conv (GEMM, M=432), f32 in -> f16 out
    gemm1x1<float, _Float16><<<dim3(HWDIM / 16, BATCH), 256, 0, stream>>>(
        x, qwh, qkv_b, y1, 432);
    // 2) depthwise 3x3 (f16 -> f16)
    dwconv3x3<<<dim3(HWDIM / 256, BATCH * 432), 256, 0, stream>>>(y1, dw_w, dw_b, y2);
    // 3) windowed channel attention, three branches (f16 -> f16)
    winattn<0, 4, 4><<<dim3(4096, BATCH), 288, s0, stream>>>(y2, ya);
    winattn<1, 8, 8><<<dim3(1024, BATCH), 288, s1, stream>>>(y2, ya);
    winattn<2, 16, 16><<<dim3(256, BATCH), 288, s2, stream>>>(y2, ya);
    // 4) proj 1x1 conv (GEMM, M=144), f16 in -> f32 out
    gemm1x1<_Float16, float><<<dim3(HWDIM / 16, BATCH), 256, 0, stream>>>(
        ya, pwh, proj_b, outp, 144);
}
